// GatedDeltaNetBlock_56831007261237
// MI455X (gfx1250) — matrix operand order repaired and checked by reading, not compiled, not run on hardware
//
#include <hip/hip_runtime.h>

// ---------------------------------------------------------------------------
// Gated DeltaNet chunkwise forward for MI455X (gfx1250, wave32, WMMA).
// B=2,H=8 -> 16 workgroups, one per (b,h). 256 threads = 8 wave32 per WG.
// State S^T (128x128 f32) lives in LDS across the 128-chunk sequential scan.
// All GEMMs run on v_wmma_f32_16x16x32_bf16; operands staged in LDS in
// layouts where every A/B fragment load is 2x ds_load_b128 (contiguous).
// ---------------------------------------------------------------------------

typedef __bf16 bf16;
typedef __attribute__((ext_vector_type(16))) __bf16 v16bf;
typedef __attribute__((ext_vector_type(8)))  __bf16 v8bf;
typedef __attribute__((ext_vector_type(8)))  float  v8f;
typedef __attribute__((ext_vector_type(4)))  float  v4f;

#define BT      64
#define KD      128
#define VD      128
#define SEQ     8192
#define NCHUNK  (SEQ / BT)
#define BH      16

// LDS leading dimensions (elements). Chosen so rows are 16B aligned and
// strided accesses don't hit a single bank.
#define ST_LD   136   // S^T f32 [128][ST_LD]
#define SOL_LD  264   // solve RHS f32 [64][SOL_LD] (256 live columns)
#define M_LD    72    // M f32 [64][M_LD]
#define BF_LD   136   // bf16 [64][BF_LD]  : kc, qn, qeg, -w
#define TR_LD   72    // bf16 [128][TR_LD] : kTr, vT ; [64][TR_LD] : Aqk

// element counts
#define NF_ST   (128 * ST_LD)
#define NF_SOL  (64 * SOL_LD)
#define NF_M    (64 * M_LD)
#define NB_BF   (64 * BF_LD)
#define NB_TR   (128 * TR_LD)
#define NB_AQK  (64 * TR_LD)
#define NF_SMALL 768   // g,gc,eg,ieg,ratio,beta,kinv,qinv,rsq...

#define SMEM_BYTES ((size_t)((NF_ST + NF_SOL + NF_M) * 4 + \
                             (4 * NB_BF + 2 * NB_TR + NB_AQK) * 2 + NF_SMALL * 4))

// --- WMMA fragment loaders --------------------------------------------------
// A matrix 16x32 bf16 (ISA 7.12.2): lane L(0..15) row M=L, half h=(L>=16).
// Per-lane K indices are two contiguous runs of 8: [c0+8h, +7], [c0+16+8h, +7].
static __device__ __forceinline__ v16bf ldsA_bf(const bf16* base, int row0, int col0,
                                                int ld, int lane) {
  int r = row0 + (lane & 15);
  int h = (lane >> 4) & 1;
  const bf16* p = base + r * ld + col0 + 8 * h;
  v8bf lo = *(const v8bf*)p;
  v8bf hi = *(const v8bf*)(p + 16);
  v16bf o;
#pragma unroll
  for (int j = 0; j < 8; ++j) { o[j] = lo[j]; o[8 + j] = hi[j]; }
  return o;
}

// B matrix 32x16 bf16: lane = column N, K runs: [c0+8h, +7], [c0+16+8h, +7].
// Source layout is [n][k] row-major (i.e. the transposed operand).
static __device__ __forceinline__ v16bf ldsB_bf(const bf16* base, int n0, int col0,
                                                int ld, int lane) {
  int r = n0 + (lane & 15);
  int h = (lane >> 4) & 1;
  const bf16* p = base + r * ld + col0 + 8 * h;
  v8bf lo = *(const v8bf*)p;
  v8bf hi = *(const v8bf*)(p + 16);
  v16bf o;
#pragma unroll
  for (int j = 0; j < 8; ++j) { o[j] = lo[j]; o[8 + j] = hi[j]; }
  return o;
}

// A fragment from an f32 LDS array (the state S^T), converted on the fly.
static __device__ __forceinline__ v16bf ldsA_f32(const float* base, int row0, int col0,
                                                 int ld, int lane) {
  int r = row0 + (lane & 15);
  int h = (lane >> 4) & 1;
  const float* p = base + r * ld + col0 + 8 * h;
  v16bf o;
#pragma unroll
  for (int j = 0; j < 8; ++j) { o[j] = (bf16)p[j]; o[8 + j] = (bf16)p[16 + j]; }
  return o;
}

static __device__ __forceinline__ v8f wmma_bf16(v16bf a, v16bf b, v8f c) {
  return __builtin_amdgcn_wmma_f32_16x16x32_bf16(false, a, false, b, (short)0, c,
                                                 false, false);
}

__global__ void __launch_bounds__(256, 1)
gdn_fwd_kernel(const float* __restrict__ q, const float* __restrict__ k,
               const float* __restrict__ v, const float* __restrict__ g,
               const float* __restrict__ beta, float* __restrict__ o_out,
               float* __restrict__ s_out) {
  extern __shared__ char smem[];
  float* ST   = (float*)smem;                 // S^T: [v][k] f32
  float* SOL  = ST + NF_ST;                   // solve RHS / scratch
  float* Ml   = SOL + NF_SOL;                 // unit-lower M
  bf16*  KC   = (bf16*)(Ml + NF_M);           // normalized k   [c][k]
  bf16*  QN   = KC + NB_BF;                   // normalized q   [c][k]
  bf16*  QEG  = QN + NB_BF;                   // q * eg         [c][k]
  bf16*  WNEG = QEG + NB_BF;                  // -w             [c][k]
  bf16*  KTR  = WNEG + NB_BF;                 // ratio*k ^T     [k][c]
  bf16*  VT   = KTR + NB_TR;                  // v_new ^T       [v][c]
  bf16*  AQK  = VT + NB_TR;                   // Aqk            [i][j]
  float* GCL  = (float*)(AQK + NB_AQK);       // cumsum g
  float* EG   = GCL + 64;                     // exp(gc)
  float* IEG  = EG + 64;                      // 1/exp(gc)
  float* RATIO= IEG + 64;                     // exp(gc[63]-gc)
  float* BL   = RATIO + 64;                   // beta
  float* KIV  = BL + 64;                      // 1/(||k||+eps)
  float* QIV  = KIV + 64;                     // scale/(||q||+eps)
  float* GRAW = QIV + 64;                     // raw g
  float* RSQ  = GRAW + 64;                    // [64][4] partial sumsq

  const int tid  = threadIdx.x;
  const int lane = tid & 31;
  const int wv   = tid >> 5;
  const int row  = tid >> 2;   // 0..63  (4 threads per sequence row)
  const int part = tid & 3;    // 0..3   (32 columns each)
  const int bh   = blockIdx.x;

  const float* qb = q + (size_t)bh * SEQ * KD;
  const float* kb = k + (size_t)bh * SEQ * KD;
  const float* vb = v + (size_t)bh * SEQ * VD;
  const float* gb = g + (size_t)bh * SEQ;
  const float* bb = beta + (size_t)bh * SEQ;

  // init state S^T = 0
  for (int e = tid; e < NF_ST; e += 256) ST[e] = 0.0f;
  __syncthreads();

  for (int n = 0; n < NCHUNK; ++n) {
    const int s0 = n * BT;

    // speculative prefetch of next chunk (global_prefetch_b8)
    if (n + 1 < NCHUNK) {
      const size_t s1 = (size_t)(n + 1) * BT;
      if (tid < 64)       __builtin_prefetch(kb + (s1 + tid) * KD, 0, 1);
      else if (tid < 128) __builtin_prefetch(qb + (s1 + (tid - 64)) * KD, 0, 1);
      else if (tid < 192) __builtin_prefetch(vb + (s1 + (tid - 128)) * VD, 0, 1);
    }

    // -------- Phase A1: stage raw k into SOL[:, :128]; load g, beta --------
    {
      const float* kr = kb + (size_t)(s0 + row) * KD + part * 32;
      float ss = 0.0f;
#pragma unroll
      for (int j = 0; j < 32; j += 4) {
        v4f x = *(const v4f*)(kr + j);
        *(v4f*)(&SOL[row * SOL_LD + part * 32 + j]) = x;
        ss += x[0] * x[0] + x[1] * x[1] + x[2] * x[2] + x[3] * x[3];
      }
      RSQ[tid] = ss;
      if (tid < 64) { GRAW[tid] = gb[s0 + tid]; BL[tid] = bb[s0 + tid]; }
    }
    __syncthreads();

    // -------- Phase A2: k row norms; gate cumsum / exp / ratio --------
    if (tid < 64) {
      float s = RSQ[tid * 4] + RSQ[tid * 4 + 1] + RSQ[tid * 4 + 2] + RSQ[tid * 4 + 3];
      KIV[tid] = 1.0f / (sqrtf(s) + 1e-6f);
      float acc = 0.0f, my = 0.0f;
#pragma unroll 8
      for (int j = 0; j < 64; ++j) { acc += GRAW[j]; if (j == tid) my = acc; }
      GCL[tid] = my;
      float e = __expf(my);
      EG[tid] = e;
      IEG[tid] = 1.0f / e;
      RATIO[tid] = __expf(acc - my);     // exp(gc[63] - gc[c])
    }
    __syncthreads();

    // -------- Phase A3: write kc, ratio*k^T; overwrite SOL with beta*eg*k;
    //          stage raw q into SOL[:, 128:256] --------
    {
      const float kiv = KIV[row];
      const float be  = BL[row] * EG[row];
      const float rr  = RATIO[row];
#pragma unroll 8
      for (int j = 0; j < 32; ++j) {
        const int c = part * 32 + j;
        const float val = SOL[row * SOL_LD + c] * kiv;
        KC[row * BF_LD + c]  = (bf16)val;
        KTR[c * TR_LD + row] = (bf16)(rr * val);
        SOL[row * SOL_LD + c] = be * val;       // rhs K-part
      }
      const float* qr = qb + (size_t)(s0 + row) * KD + part * 32;
      float ss = 0.0f;
#pragma unroll
      for (int j = 0; j < 32; j += 4) {
        v4f x = *(const v4f*)(qr + j);
        *(v4f*)(&SOL[row * SOL_LD + 128 + part * 32 + j]) = x;
        ss += x[0] * x[0] + x[1] * x[1] + x[2] * x[2] + x[3] * x[3];
      }
      RSQ[tid] = ss;
    }
    __syncthreads();

    // -------- Phase A4: q row norms (includes K^-0.5 scale) --------
    if (tid < 64) {
      float s = RSQ[tid * 4] + RSQ[tid * 4 + 1] + RSQ[tid * 4 + 2] + RSQ[tid * 4 + 3];
      QIV[tid] = 0.08838834764831845f / (sqrtf(s) + 1e-6f);   // 1/sqrt(128)
    }
    __syncthreads();

    // -------- Phase A5: write qn, q*eg; overwrite SOL[:,128:] with beta*v ---
    {
      const float qiv = QIV[row];
      const float e   = EG[row];
      const float bt  = BL[row];
#pragma unroll 8
      for (int j = 0; j < 32; ++j) {
        const int c = part * 32 + j;
        const float qv = SOL[row * SOL_LD + 128 + c] * qiv;
        QN[row * BF_LD + c]  = (bf16)qv;
        QEG[row * BF_LD + c] = (bf16)(qv * e);
      }
      const float* vr = vb + (size_t)(s0 + row) * VD + part * 32;
#pragma unroll
      for (int j = 0; j < 32; j += 4) {
        v4f x = *(const v4f*)(vr + j);
        v4f y; y[0] = bt * x[0]; y[1] = bt * x[1]; y[2] = bt * x[2]; y[3] = bt * x[3];
        *(v4f*)(&SOL[row * SOL_LD + 128 + part * 32 + j]) = y;   // rhs V-part
      }
    }
    __syncthreads();

    // -------- Phase B: M = (k k^T) .* decay .* beta ; Aqk = (q k^T) .* decay
    // 32 jobs of 16x16 tiles (16 M + 16 Aqk), 4 per wave, Kloop = 128/32. ----
    for (int job = wv * 4; job < wv * 4 + 4; ++job) {
      const bool isM = (job < 16);
      const int jt = isM ? job : job - 16;
      const int ti = jt >> 2, tj = jt & 3;
      const bf16* Abase = isM ? KC : QN;
      v8f acc;
#pragma unroll
      for (int i = 0; i < 8; ++i) acc[i] = 0.0f;
#pragma unroll
      for (int kk = 0; kk < 4; ++kk) {
        v16bf a = ldsA_bf(Abase, ti * 16, kk * 32, BF_LD, lane);
        v16bf b = ldsB_bf(KC, tj * 16, kk * 32, BF_LD, lane);
        acc = wmma_bf16(a, b, acc);
      }
      const int nn = lane & 15, h = (lane >> 4) & 1;
      const int jc = tj * 16 + nn;
#pragma unroll
      for (int i = 0; i < 8; ++i) {
        const int ir = ti * 16 + i + 8 * h;
        const float dec = EG[ir] * IEG[jc];     // exp(gc_i - gc_j)
        if (isM) {
          if (jc < ir) Ml[ir * M_LD + jc] = acc[i] * dec * BL[ir];
        } else {
          AQK[ir * TR_LD + jc] = (bf16)((jc <= ir) ? acc[i] * dec : 0.0f);
        }
      }
    }
    __syncthreads();

    // -------- Phase C: unit-lower triangular solve, one column per thread ---
    {
      const int col = tid;                       // 256 columns exactly
      for (int i = 1; i < BT; ++i) {
        float a = 0.0f;
        for (int j = 0; j < i; ++j) a += Ml[i * M_LD + j] * SOL[j * SOL_LD + col];
        SOL[i * SOL_LD + col] -= a;
      }
    }
    __syncthreads();
    // copy w (cols 0..127) negated to bf16 (bf16 WMMA has no A/B negate)
    for (int e = tid; e < 64 * 128; e += 256) {
      const int r = e >> 7, c = e & 127;
      WNEG[r * BF_LD + c] = (bf16)(-SOL[r * SOL_LD + c]);
    }
    __syncthreads();

    // -------- Phase D: v_new^T = u^T + S^T @ (-w)^T  (32 tiles, 4/wave) ----
    for (int t = 0; t < 4; ++t) {
      const int id = wv * 4 + t;
      const int tv = id >> 2, tc = id & 3;
      const int nn = lane & 15, h = (lane >> 4) & 1;
      v8f acc;
#pragma unroll
      for (int i = 0; i < 8; ++i)               // init from u (cols 128..255)
        acc[i] = SOL[(tc * 16 + nn) * SOL_LD + 128 + tv * 16 + i + 8 * h];
#pragma unroll
      for (int kk = 0; kk < 4; ++kk) {
        v16bf a = ldsA_f32(ST, tv * 16, kk * 32, ST_LD, lane);
        v16bf b = ldsB_bf(WNEG, tc * 16, kk * 32, BF_LD, lane);
        acc = wmma_bf16(a, b, acc);
      }
#pragma unroll
      for (int i = 0; i < 8; ++i)
        VT[(tv * 16 + i + 8 * h) * TR_LD + tc * 16 + nn] = (bf16)acc[i];
    }
    __syncthreads();

    // -------- Phase E: o^T = S^T @ (q*eg)^T + v_new^T @ Aqk^T; store o -----
    for (int t = 0; t < 4; ++t) {
      const int id = wv * 4 + t;
      const int tv = id >> 2, tc = id & 3;
      v8f acc;
#pragma unroll
      for (int i = 0; i < 8; ++i) acc[i] = 0.0f;
#pragma unroll
      for (int kk = 0; kk < 4; ++kk) {
        v16bf a = ldsA_f32(ST, tv * 16, kk * 32, ST_LD, lane);
        v16bf b = ldsB_bf(QEG, tc * 16, kk * 32, BF_LD, lane);
        acc = wmma_bf16(a, b, acc);
      }
#pragma unroll
      for (int cc = 0; cc < 2; ++cc) {
        v16bf a = ldsA_bf(VT, tv * 16, cc * 32, TR_LD, lane);
        v16bf b = ldsB_bf(AQK, tc * 16, cc * 32, TR_LD, lane);
        acc = wmma_bf16(a, b, acc);
      }
      const int nn = lane & 15, h = (lane >> 4) & 1;
      float* op = o_out + ((size_t)bh * SEQ + s0 + tc * 16 + nn) * VD + tv * 16 + 8 * h;
      v4f y0; y0[0] = acc[0]; y0[1] = acc[1]; y0[2] = acc[2]; y0[3] = acc[3];
      v4f y1; y1[0] = acc[4]; y1[1] = acc[5]; y1[2] = acc[6]; y1[3] = acc[7];
      *(v4f*)op = y0;
      *(v4f*)(op + 4) = y1;
    }
    __syncthreads();

    // -------- Phase F: S^T = eg_last * S^T + v_new^T @ (ratio*k) -----------
    {
      const float egl = EG[63];
      for (int t = 0; t < 8; ++t) {
        const int id = wv * 8 + t;
        const int tv = id >> 3, tk = id & 7;
        v8f acc;
#pragma unroll
        for (int i = 0; i < 8; ++i) acc[i] = 0.0f;
#pragma unroll
        for (int cc = 0; cc < 2; ++cc) {
          v16bf a = ldsA_bf(VT, tv * 16, cc * 32, TR_LD, lane);
          v16bf b = ldsB_bf(KTR, tk * 16, cc * 32, TR_LD, lane);
          acc = wmma_bf16(a, b, acc);
        }
        const int nn = lane & 15, h = (lane >> 4) & 1;
#pragma unroll
        for (int i = 0; i < 8; ++i) {
          float* p = &ST[(tv * 16 + i + 8 * h) * ST_LD + tk * 16 + nn];
          *p = *p * egl + acc[i];
        }
      }
    }
    __syncthreads();
  }

  // final state: Sf[b,h,k,v] = S^T[v][k]
  for (int e = tid; e < KD * VD; e += 256) {
    const int kk = e >> 7, vv = e & 127;
    s_out[(size_t)bh * KD * VD + e] = ST[vv * ST_LD + kk];
  }
}

extern "C" void kernel_launch(void* const* d_in, const int* in_sizes, int n_in,
                              void* d_out, int out_size, void* d_ws, size_t ws_size,
                              hipStream_t stream) {
  (void)in_sizes; (void)n_in; (void)out_size; (void)d_ws; (void)ws_size;
  const float* q    = (const float*)d_in[0];
  const float* k    = (const float*)d_in[1];
  const float* v    = (const float*)d_in[2];
  const float* g    = (const float*)d_in[3];
  const float* beta = (const float*)d_in[4];
  float* o  = (float*)d_out;                          // [B,H,S,V]
  float* sf = o + (size_t)BH * SEQ * VD;              // [B,H,K,V]
  gdn_fwd_kernel<<<dim3(BH), dim3(256), SMEM_BYTES, stream>>>(q, k, v, g, beta, o, sf);
}
